// BasicTransformerLayer_32366873543455
// MI455X (gfx1250) — compile-verified, matrix operand order checked by reading, not run on hardware
//
#include <hip/hip_runtime.h>
#include <hip/hip_bf16.h>
#include <math.h>

// ---------------- problem constants (from reference) ----------------
#define EMBED   768
#define CROSSD  512
#define HEADS   12
#define HDIM    64
#define FFN_DIM 3072
#define BATCH   8
#define SEQ     1024
#define CTXL    256
#define EPSLN   1e-5f

// ---------------- WMMA plumbing (CDNA5 gfx1250, wave32) ----------------
typedef __attribute__((ext_vector_type(16))) __bf16 v16bf;
typedef __attribute__((ext_vector_type(8)))  float  v8f;

union FragAB { v16bf v; unsigned int u[8]; };

#define WMMA_BF16(A, Bm, Cm) \
  __builtin_amdgcn_wmma_f32_16x16x32_bf16(false, (A), false, (Bm), (short)0, (Cm), false, false)

__device__ __forceinline__ unsigned short f2bf(float f) {
  union { float f; unsigned int u; } c; c.f = f;
  unsigned int u = c.u;
  u += 0x7FFFu + ((u >> 16) & 1u);   // round-to-nearest-even
  return (unsigned short)(u >> 16);
}
__device__ __forceinline__ unsigned int pack_bf2(float lo, float hi) {
  return (unsigned int)f2bf(lo) | ((unsigned int)f2bf(hi) << 16);
}
// k-offset of packed pair held in fragment dword v for a given lane-half
__device__ __forceinline__ int frag_k(int v, int half) {
  return ((v & 4) << 2) + half * 8 + (v & 3) * 2;   // (v<4?0:16) + half*8 + (v%4)*2
}
// branch-free tanh: tanh(u) = sign(u) * (1 - e^{-2|u|}) / (1 + e^{-2|u|})
__device__ __forceinline__ float tanh_fast(float u) {
  const float t = __expf(-2.0f * fabsf(u));
  return copysignf((1.0f - t) / (1.0f + t), u);
}
// CDNA5 async DMA: global -> LDS, 16B per lane, tracked by ASYNCcnt.
// lds_off: LDS byte offset == low 32 bits of the generic pointer (LDS aperture, ISA 10.2).
__device__ __forceinline__ void async_ld_b128(unsigned lds_off, const float* gaddr) {
  asm volatile("global_load_async_to_lds_b128 %0, %1, off"
               :: "v"(lds_off), "v"(gaddr) : "memory");
}

// ---------------- LayerNorm: one block per row of 768 ----------------
__global__ __launch_bounds__(256)
void ln_kernel(const float* __restrict__ x, const float* __restrict__ g,
               const float* __restrict__ b, float* __restrict__ y) {
  __shared__ float red[256];
  const int row = blockIdx.x;
  const int tid = threadIdx.x;
  const float* xr = x + (size_t)row * EMBED;
  float v0 = xr[tid], v1 = xr[tid + 256], v2 = xr[tid + 512];
  red[tid] = v0 + v1 + v2;
  __syncthreads();
  for (int o = 128; o > 0; o >>= 1) { if (tid < o) red[tid] += red[tid + o]; __syncthreads(); }
  float mu = red[0] * (1.0f / EMBED);
  __syncthreads();
  float d0 = v0 - mu, d1 = v1 - mu, d2 = v2 - mu;
  red[tid] = d0 * d0 + d1 * d1 + d2 * d2;
  __syncthreads();
  for (int o = 128; o > 0; o >>= 1) { if (tid < o) red[tid] += red[tid + o]; __syncthreads(); }
  float rstd = rsqrtf(red[0] * (1.0f / EMBED) + EPSLN);
  float* yr = y + (size_t)row * EMBED;
  yr[tid]       = d0 * rstd * g[tid]       + b[tid];
  yr[tid + 256] = d1 * rstd * g[tid + 256] + b[tid + 256];
  yr[tid + 512] = d2 * rstd * g[tid + 512] + b[tid + 512];
}

// ---------------- GEMM: C[M,N] = act(A[M,K] @ B[K,N] + bias (+ resid)) ----------------
// 64x64 block tile, K-step 32, 128 threads (4 waves in 2x2), bf16 WMMA f32-acc.
// Software-pipelined: float4 global loads for tile kb+1 issued before WMMAs of kb.
#define GT 64
#define KS 32
#define LDP 36   // u16 row stride (72 B, 8-byte aligned, conflict-free frag reads)

template<bool HAS_RESID, bool ACT_GELU>
__global__ __launch_bounds__(128)
void gemm_kernel(const float* __restrict__ A, const float* __restrict__ Bm,
                 const float* __restrict__ bias, const float* __restrict__ resid,
                 float* __restrict__ C, int M, int N, int K) {
  __shared__ unsigned short lds_a[GT][LDP];   // [m][k]
  __shared__ unsigned short lds_b[GT][LDP];   // [n][k] (B transposed)
  const int tid  = threadIdx.x;
  const int lane = tid & 31, wave = tid >> 5;
  const int half = lane >> 4, idx = lane & 15;
  const int wr = wave >> 1, wc = wave & 1;
  const int rowBase = blockIdx.y * GT;
  const int colBase = blockIdx.x * GT;

  float4 ra[4], rb[4];
  auto load_tiles = [&](int kb) {
    const int k0g = kb * KS;
#pragma unroll
    for (int i = 0; i < 4; ++i) {               // A: 64x32 f32 = 512 float4
      const int fa = tid + i * 128;
      const int r = fa >> 3, c4 = fa & 7;
      ra[i] = *(const float4*)(A + (size_t)(rowBase + r) * K + k0g + c4 * 4);
    }
#pragma unroll
    for (int i = 0; i < 4; ++i) {               // B: 32x64 f32 = 512 float4
      const int fb = tid + i * 128;
      const int r = fb >> 4, c4 = fb & 15;      // r = k, c4*4 = n (coalesced)
      rb[i] = *(const float4*)(Bm + (size_t)(k0g + r) * N + colBase + c4 * 4);
    }
  };
  auto store_tiles = [&]() {
#pragma unroll
    for (int i = 0; i < 4; ++i) {
      const int fa = tid + i * 128;
      const int r = fa >> 3, c4 = fa & 7;
      uint2 t; t.x = pack_bf2(ra[i].x, ra[i].y); t.y = pack_bf2(ra[i].z, ra[i].w);
      *(uint2*)&lds_a[r][c4 * 4] = t;           // ds_store_b64
    }
#pragma unroll
    for (int i = 0; i < 4; ++i) {
      const int fb = tid + i * 128;
      const int r = fb >> 4, c4 = fb & 15;
      lds_b[c4 * 4 + 0][r] = f2bf(rb[i].x);     // transpose on store
      lds_b[c4 * 4 + 1][r] = f2bf(rb[i].y);
      lds_b[c4 * 4 + 2][r] = f2bf(rb[i].z);
      lds_b[c4 * 4 + 3][r] = f2bf(rb[i].w);
    }
  };

  v8f acc[2][2];
  acc[0][0] = {}; acc[0][1] = {}; acc[1][0] = {}; acc[1][1] = {};

  const int nk = K / KS;
  load_tiles(0);
  store_tiles();
  __syncthreads();
  for (int kb = 0; kb < nk; ++kb) {
    const bool more = (kb + 1) < nk;
    if (more) load_tiles(kb + 1);               // overlap HBM with WMMA

    FragAB afr[2], bfr[2];
#pragma unroll
    for (int t = 0; t < 2; ++t) {
      const int am = wr * 32 + t * 16 + idx;
      const int bn = wc * 32 + t * 16 + idx;
#pragma unroll
      for (int v = 0; v < 8; ++v) {
        const int kk = frag_k(v, half);
        afr[t].u[v] = *(const unsigned int*)&lds_a[am][kk];
        bfr[t].u[v] = *(const unsigned int*)&lds_b[bn][kk];
      }
    }
#pragma unroll
    for (int tm = 0; tm < 2; ++tm)
#pragma unroll
      for (int tn = 0; tn < 2; ++tn)
        acc[tm][tn] = WMMA_BF16(afr[tm].v, bfr[tn].v, acc[tm][tn]);
    __syncthreads();
    if (more) { store_tiles(); __syncthreads(); }
  }

#pragma unroll
  for (int tm = 0; tm < 2; ++tm)
#pragma unroll
    for (int tn = 0; tn < 2; ++tn) {
      const int row0 = rowBase + wr * 32 + tm * 16 + half * 8;
      const int col  = colBase + wc * 32 + tn * 16 + idx;
      const float bcol = bias[col];
      const size_t off0 = (size_t)row0 * N + col;
#pragma unroll
      for (int v = 0; v < 8; ++v) {
        const size_t off = off0 + (size_t)v * N;
        float val = acc[tm][tn][v] + bcol;
        if (HAS_RESID) val += resid[off];
        if (ACT_GELU) {  // tanh-approx GELU (jax.nn.gelu default), branch-free
          const float x = val;
          const float u = 0.7978845608028654f * (x + 0.044715f * x * x * x);
          val = 0.5f * x * (1.0f + tanh_fast(u));
        }
        C[off] = val;
      }
    }
}

// ---------------- Flash attention (bias-added, per-head) ----------------
// grid (SEQ/64, HEADS, BATCH), 128 threads = 4 waves x 16 query rows.
// Position-bias tiles (the dominant HBM stream) are double-buffered in LDS via
// CDNA5 async DMA (global_load_async_to_lds_b128 / s_wait_asynccnt).
__global__ __launch_bounds__(128)
void attn_kernel(const float* __restrict__ Q, const float* __restrict__ Km,
                 const float* __restrict__ V, const float* __restrict__ bias,
                 float* __restrict__ O, int Tk) {
  __shared__ unsigned short lds_k[32][HDIM + 4];     // [key][d], 136 B rows
  __shared__ unsigned short lds_v[HDIM][32 + 2];     // [d][key] (transposed)
  __shared__ unsigned short lds_p[4][16][32 + 2];    // per-wave P tile, C-to-A relayout
  __shared__ float lds_bias[2][4][16][32];           // double-buffered, per-wave rows
  const int tid  = threadIdx.x;
  const int lane = tid & 31, wave = tid >> 5;
  const int half = lane >> 4, idx = lane & 15;
  const int h = blockIdx.y, b = blockIdx.z;
  const int qWave = blockIdx.x * 64 + wave * 16;     // query row within sequence
  const int hoff = h * HDIM;
  const float scale = 0.125f;                        // 1/sqrt(64)
  const size_t qBase    = (size_t)(b * SEQ + qWave) * EMBED + hoff;
  const size_t kvBase   = (size_t)b * Tk * EMBED + hoff;
  const size_t biasBase = (size_t)h * SEQ * Tk + (size_t)qWave * Tk;

  // async-issue one 16x32 bias tile for this wave's query rows (4 x b128 per lane)
  auto issue_bias = [&](int kb, int buf) {
#pragma unroll
    for (int i = 0; i < 4; ++i) {
      const int flat = i * 128 + lane * 4;           // 512 floats per wave tile
      const int row = flat >> 5, key = flat & 31;
      const float* g = bias + biasBase + (size_t)row * Tk + (size_t)kb * 32 + key;
      const unsigned lofs = (unsigned)(size_t)(const void*)&lds_bias[buf][wave][row][key];
      async_ld_b128(lofs, g);
    }
  };

  // Q fragments: d in [0,32) and [32,64)
  FragAB qf[2];
#pragma unroll
  for (int f = 0; f < 2; ++f)
#pragma unroll
    for (int v = 0; v < 8; ++v) {
      const int d = f * 32 + frag_k(v, half);
      const float* p = Q + qBase + (size_t)idx * EMBED + d;
      qf[f].u[v] = pack_bf2(p[0], p[1]);
    }

  float m_run[8], l_run[8];
  v8f oacc[4];
#pragma unroll
  for (int v = 0; v < 8; ++v) { m_run[v] = -3.0e38f; l_run[v] = 0.0f; }
  oacc[0] = {}; oacc[1] = {}; oacc[2] = {}; oacc[3] = {};

  const int nkb = Tk / 32;
  issue_bias(0, 0);
  for (int kb = 0; kb < nkb; ++kb) {
    const bool more = (kb + 1) < nkb;
    // stage K, V blocks (16 keys x 64 dims), float4 loads
#pragma unroll
    for (int i = 0; i < 2; ++i) {
      const int fl = tid + i * 128;                  // 256 float4 = 16x64 f32
      const int r = fl >> 4, c4 = fl & 15;           // r = key, c4*4 = d
      const size_t go = kvBase + (size_t)(kb * 32 + r) * EMBED + c4 * 4;
      const float4 kv = *(const float4*)(Km + go);
      uint2 t; t.x = pack_bf2(kv.x, kv.y); t.y = pack_bf2(kv.z, kv.w);
      *(uint2*)&lds_k[r][c4 * 4] = t;                // ds_store_b64
      const float4 vv = *(const float4*)(V + go);
      lds_v[c4 * 4 + 0][r] = f2bf(vv.x);             // transpose on store
      lds_v[c4 * 4 + 1][r] = f2bf(vv.y);
      lds_v[c4 * 4 + 2][r] = f2bf(vv.z);
      lds_v[c4 * 4 + 3][r] = f2bf(vv.w);
    }
    if (more) {
      issue_bias(kb + 1, (kb + 1) & 1);              // async prefetch next bias tile
      const size_t gn = kvBase + (size_t)((kb + 1) * 32 + (tid >> 2)) * EMBED + (tid & 3) * 16;
      __builtin_prefetch(Km + gn, 0, 0);             // global_prefetch for next K/V
      __builtin_prefetch(V + gn, 0, 0);
    }
    __syncthreads();

    // S = Q @ K^T : two 16-key halves, contraction over d in two 32-chunks
    v8f s0 = {}, s1 = {};
#pragma unroll
    for (int f = 0; f < 2; ++f) {
      FragAB b0, b1;
#pragma unroll
      for (int v = 0; v < 8; ++v) {
        const int d = f * 32 + frag_k(v, half);
        b0.u[v] = *(const unsigned int*)&lds_k[idx][d];
        b1.u[v] = *(const unsigned int*)&lds_k[16 + idx][d];
      }
      s0 = WMMA_BF16(qf[f].v, b0.v, s0);
      s1 = WMMA_BF16(qf[f].v, b1.v, s1);
    }

    // retire this block's async bias batch (in-order), keep the prefetch in flight
    if (more) asm volatile("s_wait_asynccnt 0x4" ::: "memory");
    else      asm volatile("s_wait_asynccnt 0x0" ::: "memory");

    // online softmax per query row (row = v + half*8, cols across 16-lane half)
    const int bb = kb & 1;
    float alpha[8];
#pragma unroll
    for (int v = 0; v < 8; ++v) {
      const int qr = v + half * 8;
      float sv0 = s0[v] * scale + lds_bias[bb][wave][qr][idx];
      float sv1 = s1[v] * scale + lds_bias[bb][wave][qr][16 + idx];
      float mx = fmaxf(sv0, sv1);
      for (int msk = 8; msk >= 1; msk >>= 1) mx = fmaxf(mx, __shfl_xor(mx, msk, 32));
      const float mnew = fmaxf(m_run[v], mx);
      const float a  = __expf(m_run[v] - mnew);
      const float p0 = __expf(sv0 - mnew);
      const float p1 = __expf(sv1 - mnew);
      float rs = p0 + p1;
      for (int msk = 8; msk >= 1; msk >>= 1) rs += __shfl_xor(rs, msk, 32);
      l_run[v] = l_run[v] * a + rs;
      m_run[v] = mnew;
      alpha[v] = a;
      lds_p[wave][qr][idx]      = f2bf(p0);
      lds_p[wave][qr][16 + idx] = f2bf(p1);
    }
#pragma unroll
    for (int t = 0; t < 4; ++t)
#pragma unroll
      for (int v = 0; v < 8; ++v) oacc[t][v] *= alpha[v];
    __syncthreads();

    // O += P @ V
    FragAB pf;
#pragma unroll
    for (int v = 0; v < 8; ++v) {
      const int kk = frag_k(v, half);
      pf.u[v] = *(const unsigned int*)&lds_p[wave][idx][kk];
    }
#pragma unroll
    for (int t = 0; t < 4; ++t) {
      FragAB vf;
#pragma unroll
      for (int v = 0; v < 8; ++v) {
        const int kk = frag_k(v, half);
        vf.u[v] = *(const unsigned int*)&lds_v[t * 16 + idx][kk];
      }
      oacc[t] = WMMA_BF16(pf.v, vf.v, oacc[t]);
    }
    __syncthreads();
  }

#pragma unroll
  for (int v = 0; v < 8; ++v) {
    const float inv = 1.0f / l_run[v];
    const size_t ro = (size_t)(b * SEQ + qWave + v + half * 8) * EMBED + hoff;
#pragma unroll
    for (int t = 0; t < 4; ++t) O[ro + t * 16 + idx] = oacc[t][v] * inv;
  }
}

// ---------------- host orchestration ----------------
extern "C" void kernel_launch(void* const* d_in, const int* in_sizes, int n_in,
                              void* d_out, int out_size, void* d_ws, size_t ws_size,
                              hipStream_t stream) {
  const float* hidden  = (const float*)d_in[0];
  const float* context = (const float*)d_in[1];
  const float* cn_g = (const float*)d_in[2];  const float* cn_b = (const float*)d_in[3];
  const float* wq_c = (const float*)d_in[4];  const float* bq_c = (const float*)d_in[5];
  const float* wk_c = (const float*)d_in[6];  const float* bk_c = (const float*)d_in[7];
  const float* wv_c = (const float*)d_in[8];  const float* bv_c = (const float*)d_in[9];
  const float* wo_c = (const float*)d_in[10]; const float* bo_c = (const float*)d_in[11];
  const float* bias_c = (const float*)d_in[12];
  const float* sn_g = (const float*)d_in[13]; const float* sn_b = (const float*)d_in[14];
  const float* wq_s = (const float*)d_in[15]; const float* bq_s = (const float*)d_in[16];
  const float* wk_s = (const float*)d_in[17]; const float* bk_s = (const float*)d_in[18];
  const float* wv_s = (const float*)d_in[19]; const float* bv_s = (const float*)d_in[20];
  const float* wo_s = (const float*)d_in[21]; const float* bo_s = (const float*)d_in[22];
  const float* bias_s = (const float*)d_in[23];
  const float* fn_g = (const float*)d_in[24]; const float* fn_b = (const float*)d_in[25];
  const float* w1 = (const float*)d_in[26];   const float* b1 = (const float*)d_in[27];
  const float* w2 = (const float*)d_in[28];   const float* b2 = (const float*)d_in[29];
  float* out = (float*)d_out;

  const size_t NE = (size_t)BATCH * SEQ * EMBED;     // 6,291,456 floats
  float* ws    = (float*)d_ws;
  float* x_ln  = ws;              // size NE
  float* h1    = ws + NE;         // size NE
  float* h2    = ws + 2 * NE;     // size NE
  float* qbuf  = ws + 3 * NE;     // size NE  -- region [3NE,7NE) reused as FFN mid
  float* kbuf  = ws + 4 * NE;     // size NE
  float* vbuf  = ws + 5 * NE;     // size NE
  float* abuf  = ws + 6 * NE;     // size NE
  float* ffn   = ws + 3 * NE;     // size 4*NE == BATCH*SEQ*FFN_DIM

  const int MS = BATCH * SEQ;    // 8192
  const int ML = BATCH * CTXL;   // 2048
  const dim3 blk128(128), blk256(256);
  const dim3 gAttn(SEQ / 64, HEADS, BATCH);
#define GEMM_GRID(M, N) dim3((N) / GT, (M) / GT)

  // ---- cross attention block ----
  ln_kernel<<<MS, blk256, 0, stream>>>(hidden, cn_g, cn_b, x_ln);
  gemm_kernel<false,false><<<GEMM_GRID(MS, EMBED), blk128, 0, stream>>>(x_ln, wq_c, bq_c, nullptr, qbuf, MS, EMBED, EMBED);
  gemm_kernel<false,false><<<GEMM_GRID(ML, EMBED), blk128, 0, stream>>>(context, wk_c, bk_c, nullptr, kbuf, ML, EMBED, CROSSD);
  gemm_kernel<false,false><<<GEMM_GRID(ML, EMBED), blk128, 0, stream>>>(context, wv_c, bv_c, nullptr, vbuf, ML, EMBED, CROSSD);
  attn_kernel<<<gAttn, blk128, 0, stream>>>(qbuf, kbuf, vbuf, bias_c, abuf, CTXL);
  gemm_kernel<true,false><<<GEMM_GRID(MS, EMBED), blk128, 0, stream>>>(abuf, wo_c, bo_c, hidden, h1, MS, EMBED, EMBED);

  // ---- self attention block ----
  ln_kernel<<<MS, blk256, 0, stream>>>(h1, sn_g, sn_b, x_ln);
  gemm_kernel<false,false><<<GEMM_GRID(MS, EMBED), blk128, 0, stream>>>(x_ln, wq_s, bq_s, nullptr, qbuf, MS, EMBED, EMBED);
  gemm_kernel<false,false><<<GEMM_GRID(MS, EMBED), blk128, 0, stream>>>(x_ln, wk_s, bk_s, nullptr, kbuf, MS, EMBED, EMBED);
  gemm_kernel<false,false><<<GEMM_GRID(MS, EMBED), blk128, 0, stream>>>(x_ln, wv_s, bv_s, nullptr, vbuf, MS, EMBED, EMBED);
  attn_kernel<<<gAttn, blk128, 0, stream>>>(qbuf, kbuf, vbuf, bias_s, abuf, SEQ);
  gemm_kernel<true,false><<<GEMM_GRID(MS, EMBED), blk128, 0, stream>>>(abuf, wo_s, bo_s, h1, h2, MS, EMBED, EMBED);

  // ---- FFN (no residual per reference) ----
  ln_kernel<<<MS, blk256, 0, stream>>>(h2, fn_g, fn_b, x_ln);
  gemm_kernel<false,true><<<GEMM_GRID(MS, FFN_DIM), blk128, 0, stream>>>(x_ln, w1, b1, nullptr, ffn, MS, FFN_DIM, EMBED);
  gemm_kernel<false,false><<<GEMM_GRID(MS, EMBED), blk128, 0, stream>>>(ffn, w2, b2, nullptr, out, MS, EMBED, FFN_DIM);
}
